// MultiHeadAttention_27857157881860
// MI455X (gfx1250) — matrix operand order repaired and checked by reading, not compiled, not run on hardware
//
#include <hip/hip_runtime.h>
#include <stdint.h>

#define WIDTH 1024
#define HEADS 16
#define HD    64
#define NB    4
#define SEQ   2048

typedef float  v8f   __attribute__((ext_vector_type(8)));
typedef __bf16 v16bf __attribute__((ext_vector_type(16)));
typedef unsigned int u32x4 __attribute__((ext_vector_type(4)));
typedef int          i32x8 __attribute__((ext_vector_type(8)));

union BF16Frag {
    uint4          q[2];
    unsigned short s[16];
    v16bf          v;
};

__device__ __forceinline__ unsigned short f2bf(float f) {
    unsigned u = __float_as_uint(f);
    unsigned r = (u + 0x7FFFu + ((u >> 16) & 1u)) >> 16;   // round-to-nearest-even
    return (unsigned short)r;
}

__device__ __forceinline__ v8f v8f_zero() {
    v8f z = {0.f, 0.f, 0.f, 0.f, 0.f, 0.f, 0.f, 0.f};
    return z;
}

__device__ __forceinline__ v8f wmma_bf16(v16bf a, v16bf b, v8f c) {
    return __builtin_amdgcn_wmma_f32_16x16x32_bf16(false, a, false, b, (short)0, c, false, false);
}

// ---------------------------------------------------------------------------
// TDM: 1-D DMA of (n8 * 8) bytes global -> LDS.  Descriptor per CDNA5 ISA §8:
//  group0: count=1 | lds_addr | global_addr(57b) | type=2("image")
//  group1: data_size=3 (8B units), tensor_dim0 = tile_dim0 = n8, dims1+=1.
// lds_off = low 32 bits of flat LDS pointer (= LDS byte offset per aperture rule).
// Issued by one wave; completion tracked with TENSORcnt.
// ---------------------------------------------------------------------------
__device__ __forceinline__ void tdm_load_1d(const void* gptr, unsigned lds_off, unsigned n8) {
    unsigned long long ga = (unsigned long long)(uintptr_t)gptr;
    u32x4 g0;
    g0.x = 1u;                                            // count=1, user mode
    g0.y = lds_off;                                       // lds_addr
    g0.z = (unsigned)ga;                                  // global_addr[31:0]
    g0.w = (unsigned)((ga >> 32) & 0x01FFFFFFu) | (2u << 30);  // addr[56:32] | type=2
    i32x8 g1;
    g1[0] = 0x00030000;                                   // data_size=3 (8 bytes)
    g1[1] = (int)((n8 & 0xFFFFu) << 16);                  // tensor_dim0 lo
    g1[2] = (int)(((n8 >> 16) & 0xFFFFu) | (1u << 16));   // tensor_dim0 hi | tensor_dim1=1
    g1[3] = (int)((n8 & 0xFFFFu) << 16);                  // tile_dim0
    g1[4] = 1;                                            // tile_dim1=1
    g1[5] = (int)n8;                                      // tensor_dim0_stride
    g1[6] = 0;
    g1[7] = 0;
    asm volatile("tensor_load_to_lds %0, %1" :: "s"(g0), "s"(g1) : "memory");
}

// ---------------------------------------------------------------------------
// Kernel W: convert W[n][k] fp32 -> bf16, scattered into WMMA B-fragment tiling:
//   frag[ko][nt][lane][p], ko = k/32, nt = n/16, lane = (n&15) + ((k%32)/16)*16,
//   p = k&15.  B[k][n] = W[n][k]  (since GEMMs compute x @ W^T).
// ---------------------------------------------------------------------------
__global__ __launch_bounds__(256) void weights_to_bfrag(
    const float* __restrict__ W0, const float* __restrict__ W1,
    const float* __restrict__ W2, const float* __restrict__ W3,
    unsigned short* __restrict__ out)
{
    int idx = blockIdx.x * 256 + threadIdx.x;   // 4 * 1024 * 128 threads
    int mat = idx >> 17;
    int rem = idx & 131071;
    int n   = rem >> 7;
    int k   = (rem & 127) << 3;
    const float* W = (mat == 0) ? W0 : (mat == 1) ? W1 : (mat == 2) ? W2 : W3;

    float4 f0 = *(const float4*)(W + (size_t)n * WIDTH + k);
    float4 f1 = *(const float4*)(W + (size_t)n * WIDTH + k + 4);

    union { unsigned short s[8]; uint4 q; } pk;
    pk.s[0] = f2bf(f0.x); pk.s[1] = f2bf(f0.y); pk.s[2] = f2bf(f0.z); pk.s[3] = f2bf(f0.w);
    pk.s[4] = f2bf(f1.x); pk.s[5] = f2bf(f1.y); pk.s[6] = f2bf(f1.z); pk.s[7] = f2bf(f1.w);

    int ko   = k >> 5;
    int kk   = k & 31;
    int lane = (n & 15) + ((kk >> 4) << 4);
    int p    = kk & 15;
    int nt   = n >> 4;
    size_t base = (size_t)mat * 1048576 + ((((size_t)ko * 64) + nt) * 32 + lane) * 16 + p;
    *(uint4*)(out + base) = pk.q;
}

// ---------------------------------------------------------------------------
// GEMM: out[m,n] = act[m,:] . W[n,:]   (M=8192, N=1024, K=1024)
// B operand: TDM-staged into double-buffered LDS (8KB contiguous per K-step,
// thanks to the fragment-tiled weight layout), drained with s_wait_tensorcnt.
// ACT_BF16: 0 = fp32 activations (convert on load), 1 = bf16 activations.
// OUT_MODE: 0 = bf16 row-major, 1 = fp32 row-major, 2 = bf16 V-transpose.
// ---------------------------------------------------------------------------
template <int ACT_BF16, int OUT_MODE>
__global__ __launch_bounds__(256) void gemm_wmma(
    const void* __restrict__ act,
    const unsigned short* __restrict__ wfrag,
    void* __restrict__ out)
{
    __shared__ unsigned short bstage[2][4096];   // 2 x 8KB B-fragment slabs

    const int tid  = threadIdx.x;
    const int lane = tid & 31;
    const int wid  = tid >> 5;
    const int wm   = wid & 3;
    const int wn   = wid >> 2;
    const int m0   = blockIdx.x * 128 + wm * 32;
    const int nblk = blockIdx.y;
    const int l15  = lane & 15;
    const int c0   = (lane < 16) ? 0 : 16;
    const unsigned lds0 = (unsigned)(uintptr_t)&bstage[0][0];

    v8f acc[2][4];
#pragma unroll
    for (int mt = 0; mt < 2; ++mt)
#pragma unroll
        for (int nt = 0; nt < 4; ++nt) acc[mt][nt] = v8f_zero();

    // prologue: DMA B slab for ko=0
    if (wid == 0)
        tdm_load_1d(wfrag + (size_t)nblk * 8 * 512, lds0, 1024);

    for (int ko = 0; ko < 32; ++ko) {
        const int k0 = ko * 32;
        __builtin_amdgcn_s_wait_tensorcnt(0);     // slab for this ko landed
        __syncthreads();
        if (ko + 1 < 32 && wid == 0)              // overlap next DMA with compute
            tdm_load_1d(wfrag + (size_t)((ko + 1) * 64 + nblk * 8) * 512,
                        lds0 + (unsigned)(((ko + 1) & 1) * 8192), 1024);

        BF16Frag a[2];
#pragma unroll
        for (int mt = 0; mt < 2; ++mt) {
            const int m = m0 + mt * 16 + l15;
            if constexpr (ACT_BF16) {
                const unsigned short* p0 = (const unsigned short*)act + (size_t)m * WIDTH + k0 + c0;
                a[mt].q[0] = *(const uint4*)p0;
                a[mt].q[1] = *(const uint4*)(p0 + 8);
            } else {
                const float* p0 = (const float*)act + (size_t)m * WIDTH + k0 + c0;
                float4 f0 = *(const float4*)(p0);
                float4 f1 = *(const float4*)(p0 + 4);
                float4 f2 = *(const float4*)(p0 + 8);
                float4 f3 = *(const float4*)(p0 + 12);
                const float ff[16] = {f0.x, f0.y, f0.z, f0.w, f1.x, f1.y, f1.z, f1.w,
                                      f2.x, f2.y, f2.z, f2.w, f3.x, f3.y, f3.z, f3.w};
#pragma unroll
                for (int e = 0; e < 16; ++e) a[mt].s[e] = f2bf(ff[e]);
            }
        }

        const unsigned short* bbuf = &bstage[ko & 1][0];
#pragma unroll
        for (int nt = 0; nt < 4; ++nt) {
            const unsigned short* bp = bbuf + (((wn * 4 + nt) * 32) + lane) * 16;
            BF16Frag b;
            b.q[0] = *(const uint4*)bp;
            b.q[1] = *(const uint4*)(bp + 8);
#pragma unroll
            for (int mt = 0; mt < 2; ++mt)
                acc[mt][nt] = wmma_bf16(a[mt].v, b.v, acc[mt][nt]);
        }
        __syncthreads();                          // slab free before DMA reuses it
    }

    const int rbase = (lane >> 4) * 8;
    const int n0 = nblk * 128 + wn * 64;
#pragma unroll
    for (int mt = 0; mt < 2; ++mt)
#pragma unroll
        for (int nt = 0; nt < 4; ++nt)
#pragma unroll
            for (int r = 0; r < 8; ++r) {
                const int row = m0 + mt * 16 + r + rbase;
                const int col = n0 + nt * 16 + l15;
                const float val = acc[mt][nt][r];
                if constexpr (OUT_MODE == 0) {
                    ((unsigned short*)out)[(size_t)row * WIDTH + col] = f2bf(val);
                } else if constexpr (OUT_MODE == 1) {
                    ((float*)out)[(size_t)row * WIDTH + col] = val;
                } else {
                    const int bb   = row >> 11;
                    const int jrow = row & (SEQ - 1);
                    const int h    = jrow >> 7;
                    const int j2   = ((jrow & 127) << 4) + (col >> 6);
                    const int d    = col & 63;
                    const size_t bh = (size_t)bb * HEADS + h;
                    ((unsigned short*)out)[((bh * HD + d) << 11) + j2] = f2bf(val);
                }
            }
}

// ---------------------------------------------------------------------------
// Fused attention per (bh, 16-row i-block): scores in VGPRs, cross-wave
// softmax, required fp32 attn write, PV with per-wave LDS restage (wave-local,
// so only wave_barrier needed), X reduced via LDS float atomics.
// ---------------------------------------------------------------------------
__global__ __launch_bounds__(256) void attention_wmma(
    const unsigned short* __restrict__ qp,
    const unsigned short* __restrict__ kp,
    const unsigned short* __restrict__ vpt,
    const unsigned char* __restrict__ mask,
    float* __restrict__ attn,
    unsigned short* __restrict__ xp)
{
    __shared__ float smax[8 * 16];
    __shared__ float ssum[8 * 16];
    __shared__ float gmax[16];
    __shared__ float ginv[16];
    __shared__ float xbuf[16 * 64];
    __shared__ unsigned short pstage[8][16 * 32];

    const int tid  = threadIdx.x;
    const int lane = tid & 31;
    const int wid  = tid >> 5;
    const int l15  = lane & 15;
    const int g    = lane >> 4;
    const int c0   = g ? 16 : 0;
    const int i0   = blockIdx.x * 16;
    const int bh   = blockIdx.y;
    const int b    = bh >> 4;
    const int h    = bh & 15;
    const size_t hb = (size_t)bh * (SEQ * HD);
    const int jw   = wid * 256;

    for (int e = tid; e < 16 * 64; e += 256) xbuf[e] = 0.f;

    BF16Frag aq[2];
    {
        const unsigned short* qrow = qp + hb + (size_t)(i0 + l15) * HD;
#pragma unroll
        for (int ks = 0; ks < 2; ++ks) {
            const unsigned short* p0 = qrow + ks * 32 + c0;
            aq[ks].q[0] = *(const uint4*)p0;
            aq[ks].q[1] = *(const uint4*)(p0 + 8);
        }
    }

    // ---- scores ----
    v8f sacc[16];
#pragma unroll
    for (int nt = 0; nt < 16; ++nt) {
        const int j0 = jw + nt * 16;
        v8f c = v8f_zero();
#pragma unroll
        for (int ks = 0; ks < 2; ++ks) {
            const unsigned short* kr = kp + hb + (size_t)(j0 + l15) * HD + ks * 32 + g * 16;
            BF16Frag bf;
            bf.q[0] = *(const uint4*)kr;
            bf.q[1] = *(const uint4*)(kr + 8);
            c = wmma_bf16(aq[ks].v, bf.v, c);
        }
        const int j = j0 + l15;
        const unsigned char* mp = mask + ((size_t)b * SEQ + i0 + g * 8) * SEQ + j;
#pragma unroll
        for (int r = 0; r < 8; ++r) {
            float s = c[r];
            s = mp[(size_t)r * SEQ] ? -1e9f : s;
            c[r] = s * 0.125f;
        }
        sacc[nt] = c;
    }

    // ---- softmax ----
    float rmax[8];
#pragma unroll
    for (int r = 0; r < 8; ++r) {
        float m = sacc[0][r];
#pragma unroll
        for (int nt = 1; nt < 16; ++nt) m = fmaxf(m, sacc[nt][r]);
        rmax[r] = m;
    }
#pragma unroll
    for (int off = 8; off >= 1; off >>= 1)
#pragma unroll
        for (int r = 0; r < 8; ++r) rmax[r] = fmaxf(rmax[r], __shfl_xor(rmax[r], off, 32));
    if (l15 == 0) {
#pragma unroll
        for (int r = 0; r < 8; ++r) smax[wid * 16 + g * 8 + r] = rmax[r];
    }
    __syncthreads();
    if (tid < 16) {
        float m = smax[tid];
        for (int w = 1; w < 8; ++w) m = fmaxf(m, smax[w * 16 + tid]);
        gmax[tid] = m;
    }
    __syncthreads();

    float rsum[8];
#pragma unroll
    for (int r = 0; r < 8; ++r) rsum[r] = 0.f;
#pragma unroll
    for (int nt = 0; nt < 16; ++nt)
#pragma unroll
        for (int r = 0; r < 8; ++r) {
            float e = __expf(sacc[nt][r] - gmax[g * 8 + r]);
            sacc[nt][r] = e;
            rsum[r] += e;
        }
#pragma unroll
    for (int off = 8; off >= 1; off >>= 1)
#pragma unroll
        for (int r = 0; r < 8; ++r) rsum[r] += __shfl_xor(rsum[r], off, 32);
    if (l15 == 0) {
#pragma unroll
        for (int r = 0; r < 8; ++r) ssum[wid * 16 + g * 8 + r] = rsum[r];
    }
    __syncthreads();
    if (tid < 16) {
        float s = 0.f;
        for (int w = 0; w < 8; ++w) s += ssum[w * 16 + tid];
        ginv[tid] = 1.0f / s;
    }
    __syncthreads();

    // normalize + required attn_weight output
#pragma unroll
    for (int nt = 0; nt < 16; ++nt) {
        const int j = jw + nt * 16 + l15;
        float* arow = attn + ((size_t)bh * SEQ + i0 + g * 8) * SEQ + j;
#pragma unroll
        for (int r = 0; r < 8; ++r) {
            const float p = sacc[nt][r] * ginv[g * 8 + r];
            sacc[nt][r] = p;
            arow[(size_t)r * SEQ] = p;
        }
    }

    // ---- PV ----
    v8f xacc[4];
#pragma unroll
    for (int nt = 0; nt < 4; ++nt) xacc[nt] = v8f_zero();

    unsigned short* wsb = pstage[wid];
    for (int ksj = 0; ksj < 8; ++ksj) {
#pragma unroll
        for (int t = 0; t < 2; ++t) {
            const int nt = ksj * 2 + t;
#pragma unroll
            for (int r = 0; r < 8; ++r)
                wsb[(r + g * 8) * 32 + t * 16 + l15] = f2bf(sacc[nt][r]);
        }
        __builtin_amdgcn_wave_barrier();   // wave-local LDS: DS in-order per wave

        BF16Frag ap;
        {
            const unsigned short* pr = wsb + l15 * 32 + c0;
            ap.q[0] = *(const uint4*)pr;
            ap.q[1] = *(const uint4*)(pr + 8);
        }
#pragma unroll
        for (int nt = 0; nt < 4; ++nt) {
            const unsigned short* vr =
                vpt + hb + (size_t)(nt * 16 + l15) * SEQ + jw + ksj * 32 + g * 16;
            BF16Frag bv;
            bv.q[0] = *(const uint4*)vr;
            bv.q[1] = *(const uint4*)(vr + 8);
            xacc[nt] = wmma_bf16(ap.v, bv.v, xacc[nt]);
        }
        __builtin_amdgcn_wave_barrier();
    }

#pragma unroll
    for (int nt = 0; nt < 4; ++nt)
#pragma unroll
        for (int r = 0; r < 8; ++r)
            atomicAdd(&xbuf[(r + g * 8) * 64 + nt * 16 + l15], xacc[nt][r]);
    __syncthreads();

    for (int e = tid; e < 16 * 64; e += 256) {
        const int row = e >> 6;
        const int d   = e & 63;
        xp[((size_t)b * SEQ + i0 + row) * WIDTH + h * HD + d] = f2bf(xbuf[e]);
    }
}

// ---------------------------------------------------------------------------
// Launch graph. Workspace (72 MB): [0,8M) weight frags, [8M,24M) Qp,
// [24M,40M) Kp, [40M,56M) VpT, [56M,72M) Xp.
// ---------------------------------------------------------------------------
extern "C" void kernel_launch(void* const* d_in, const int* in_sizes, int n_in,
                              void* d_out, int out_size, void* d_ws, size_t ws_size,
                              hipStream_t stream)
{
    const float* q  = (const float*)d_in[0];
    const float* k  = (const float*)d_in[1];
    const float* v  = (const float*)d_in[2];
    const unsigned char* mask = (const unsigned char*)d_in[3];
    const float* Wq = (const float*)d_in[4];
    const float* Wk = (const float*)d_in[5];
    const float* Wv = (const float*)d_in[6];
    const float* Wo = (const float*)d_in[7];

    char* ws = (char*)d_ws;
    unsigned short* wfrag = (unsigned short*)ws;
    unsigned short* qp    = (unsigned short*)(ws + ((size_t)8  << 20));
    unsigned short* kp    = (unsigned short*)(ws + ((size_t)24 << 20));
    unsigned short* vpt   = (unsigned short*)(ws + ((size_t)40 << 20));
    unsigned short* xp    = (unsigned short*)(ws + ((size_t)56 << 20));

    float* out  = (float*)d_out;
    float* attn = out + (size_t)NB * SEQ * WIDTH;

    weights_to_bfrag<<<2048, 256, 0, stream>>>(Wq, Wk, Wv, Wo, wfrag);

    dim3 gg(64, 8);
    gemm_wmma<0, 0><<<gg, 256, 0, stream>>>(q, wfrag + (size_t)0 * 1048576, qp);
    gemm_wmma<0, 0><<<gg, 256, 0, stream>>>(k, wfrag + (size_t)1 * 1048576, kp);
    gemm_wmma<0, 2><<<gg, 256, 0, stream>>>(v, wfrag + (size_t)2 * 1048576, vpt);

    attention_wmma<<<dim3(SEQ / 16, NB * HEADS), 256, 0, stream>>>(qp, kp, vpt, mask, attn, xp);

    gemm_wmma<1, 1><<<gg, 256, 0, stream>>>(xp, wfrag + (size_t)3 * 1048576, out);
}
